// MultiHeadBlock_55817394979576
// MI455X (gfx1250) — matrix operand order repaired and checked by reading, not compiled, not run on hardware
//
#include <hip/hip_runtime.h>
#include <hip/hip_bf16.h>

// ---------------------------------------------------------------------------
// MI455X (gfx1250) multi-head attention block, bf16 WMMA + async-LDS staging.
//   B=2, S=2048, D=1024, H=16, DK=64
// ---------------------------------------------------------------------------

typedef __attribute__((ext_vector_type(16))) __bf16 v16bf;
typedef __attribute__((ext_vector_type(8)))  float  v8f;

union FragAB { v16bf v; uint4 q[2]; };          // 16 bf16 = 8 VGPRs
union FragCD { v8f   v; float f[8]; };          // 8 f32   = 8 VGPRs

static __device__ __forceinline__ unsigned short f2bf(float f) {
    unsigned int u = __builtin_bit_cast(unsigned int, f);
    u += 0x7fffu + ((u >> 16) & 1u);            // round-to-nearest-even
    return (unsigned short)(u >> 16);
}

static __device__ __forceinline__ v8f wmma_bf16(const FragAB& a, const FragAB& b, v8f c) {
    return __builtin_amdgcn_wmma_f32_16x16x32_bf16(
        /*neg_a=*/false, a.v, /*neg_b=*/false, b.v,
        /*c_mod=*/(short)0, c, /*reuse_a=*/false, /*reuse_b=*/false);
}

// Async global -> LDS copy, 16 bytes per lane (ASYNCcnt-tracked).
static __device__ __forceinline__ void async_g2l_b128(void* lds, const void* g) {
    const unsigned lds_off = (unsigned)(unsigned long long)lds;  // LDS aperture: addr[31:0]
    asm volatile("global_load_async_to_lds_b128 %0, %1, off"
                 :: "v"(lds_off), "v"(g)
                 : "memory");
}
static __device__ __forceinline__ void wait_async0() {
    asm volatile("s_wait_asynccnt 0" ::: "memory");
}

static constexpr int Bc = 2, Sc = 2048, Dc = 1024, Hc = 16, DKc = 64;

// ---------------------------------------------------------------------------
// Generic GEMM:  out[M,N] = A[M,K] * W[N,K]^T + bias[N]
//   A: fp32 or bf16(raw ushort); W,bias: fp32; out: bf16 or fp32.
//   Workgroup = 256 threads (8 waves), tile 128(M) x 128(N), K staged 32-wide.
// ---------------------------------------------------------------------------
template<typename AT, bool OUTF32>
__global__ __launch_bounds__(256)
void gemm_wmma(const AT* __restrict__ A, const float* __restrict__ W,
               const float* __restrict__ bias, void* __restrict__ out,
               int M, int N, int K)
{
    __shared__ __align__(16) unsigned short As[128 * 32];       // [m][k] bf16
    __shared__ __align__(16) unsigned short Wt[32 * 136];       // [n][k] bf16 (row stride 32, not padded)

    const int tid  = threadIdx.x;
    const int w    = tid >> 5;          // wave id 0..7
    const int l    = tid & 31;          // lane
    const int half = l >> 4;            // 0: lanes 0-15, 1: lanes 16-31
    const int lm   = l & 15;

    const int Mbase = blockIdx.x * 128;
    const int Nbase = blockIdx.y * 128;

    FragCD acc[8];
#pragma unroll
    for (int j = 0; j < 8; ++j)
#pragma unroll
        for (int g = 0; g < 8; ++g) acc[j].f[g] = 0.0f;

    for (int k0 = 0; k0 < K; k0 += 32) {
        __syncthreads();                        // protect previous-iter LDS reads
        // ---- stage A tile (128 x 32) as bf16, coalesced along K ----
#pragma unroll
        for (int e = tid; e < 128 * 32; e += 256) {
            const int r = e >> 5, kk = e & 31;
            if constexpr (sizeof(AT) == 4) {
                As[e] = f2bf(((const float*)A)[(size_t)(Mbase + r) * K + k0 + kk]);
            } else {
                As[e] = ((const unsigned short*)A)[(size_t)(Mbase + r) * K + k0 + kk];
            }
        }
        // ---- stage W tile: Wt[n][k] = W[Nbase+n][k0+k] ----
#pragma unroll
        for (int e = tid; e < 128 * 32; e += 256) {
            const int n = e >> 5, kk = e & 31;
            Wt[n * 32 + kk] = f2bf(W[(size_t)(Nbase + n) * K + k0 + kk]);
        }
        // ---- prefetch next k-tile rows while this tile is consumed ----
        if (k0 + 32 < K && tid < 128) {
            __builtin_prefetch(&A[(size_t)(Mbase + tid) * K + k0 + 32], 0, 3);
            __builtin_prefetch(&W[(size_t)(Nbase + tid) * K + k0 + 32], 0, 3);
        }
        __syncthreads();

        // ---- A fragment: 16x32, lane<16 holds row M=lane, K={0..7,16..23} ----
        FragAB af;
        const unsigned short* arow = &As[(w * 16 + lm) * 32];
        af.q[0] = *(const uint4*)(arow + 8 * half);
        af.q[1] = *(const uint4*)(arow + 16 + 8 * half);

#pragma unroll
        for (int j = 0; j < 8; ++j) {
            // B fragment: 32x16, lane holds col N=lm, K={0..7,16..23} (+8 for half 1)
            FragAB bf_;
            const unsigned short* brow = &Wt[(j * 16 + lm) * 32];
            bf_.q[0] = *(const uint4*)(brow + 8 * half);
            bf_.q[1] = *(const uint4*)(brow + 16 + 8 * half);
            acc[j].v = wmma_bf16(af, bf_, acc[j].v);
        }
    }

    // ---- epilogue: C/D layout row = g + 8*half, col = lm ----
#pragma unroll
    for (int j = 0; j < 8; ++j) {
        const int c  = Nbase + j * 16 + lm;
        const float bv = bias[c];
#pragma unroll
        for (int g = 0; g < 8; ++g) {
            const int r = Mbase + w * 16 + g + 8 * half;
            const float val = acc[j].f[g] + bv;
            if constexpr (OUTF32)
                ((float*)out)[(size_t)r * N + c] = val;
            else
                ((unsigned short*)out)[(size_t)r * N + c] = f2bf(val);
        }
    }
}

// ---------------------------------------------------------------------------
// Causal flash attention over bf16 Q/K/V in [B,S,H*DK] layout.
//   grid.x = S/128 query tiles, grid.y = B*H. 256 threads = 8 waves,
//   wave w owns query rows [qbase + 16w, qbase + 16w + 16).
//   Q/K/V tiles staged with GLOBAL_LOAD_ASYNC_TO_LDS_B128 (ASYNCcnt).
// ---------------------------------------------------------------------------
__global__ __launch_bounds__(256)
void attn_wmma(const unsigned short* __restrict__ Qp,
               const unsigned short* __restrict__ Kp,
               const unsigned short* __restrict__ Vp,
               unsigned short* __restrict__ ctx)
{
    __shared__ __align__(16) unsigned short Qs  [128 * 64];     // [q][dk]
    __shared__ __align__(16) unsigned short Kraw[64 * 64];      // [key][dk] (async landing)
    __shared__ __align__(16) unsigned short Kts [64 * 72];      // [dk][key] (padded)
    __shared__ __align__(16) unsigned short Vs  [64 * 64];      // [key][dk]
    __shared__ __align__(16) unsigned short Ps  [8 * 16 * 64];  // per-wave P [row][key]

    const int tid  = threadIdx.x;
    const int w    = tid >> 5;
    const int l    = tid & 31;
    const int half = l >> 4;
    const int lm   = l & 15;

    const int qtile = blockIdx.x;
    const int bh    = blockIdx.y;
    const int b     = bh >> 4;
    const int h     = bh & 15;
    const int qbase = qtile * 128;

    const size_t head_off = ((size_t)b * Sc) * Dc + (size_t)h * DKc;

    // ---- async-stage the 128x64 Q tile (16 KB, 4 rounds of 16B/lane) ----
    {
        const unsigned short* gQ = Qp + head_off + (size_t)qbase * Dc;
#pragma unroll
        for (int rnd = 0; rnd < 4; ++rnd) {
            const int e8 = tid + rnd * 256;             // 16-byte chunk id
            const int r = e8 >> 3, c8 = (e8 & 7) * 8;
            async_g2l_b128(&Qs[r * 64 + c8], gQ + (size_t)r * Dc + c8);
        }
    }

    FragCD o[4];
    float rmax[8], rsum[8];
#pragma unroll
    for (int g = 0; g < 8; ++g) { rmax[g] = -3.0e38f; rsum[g] = 0.0f; }
#pragma unroll
    for (int jd = 0; jd < 4; ++jd)
#pragma unroll
        for (int g = 0; g < 8; ++g) o[jd].f[g] = 0.0f;

    const int nkt = 2 * qtile + 2;          // causal: key tiles up to diagonal
    for (int kt = 0; kt < nkt; ++kt) {
        const int key0 = kt * 64;
        const unsigned short* gK = Kp + head_off + (size_t)key0 * Dc;
        const unsigned short* gV = Vp + head_off + (size_t)key0 * Dc;

        __syncthreads();                    // previous-iter Kraw/Vs/Ps reads done
        // ---- async-stage K (row-major landing) and V tiles ----
#pragma unroll
        for (int rnd = 0; rnd < 2; ++rnd) {
            const int e8 = tid + rnd * 256;
            const int key = e8 >> 3, c8 = (e8 & 7) * 8;
            async_g2l_b128(&Kraw[key * 64 + c8], gK + (size_t)key * Dc + c8);
            async_g2l_b128(&Vs[key * 64 + c8],   gV + (size_t)key * Dc + c8);
        }
        wait_async0();                      // this wave's async copies (incl. Q) landed
        __syncthreads();                    // all waves' copies landed

        // ---- LDS transpose Kraw[key][dk] -> Kts[dk][key] ----
#pragma unroll
        for (int e = tid; e < 64 * 64; e += 256) {
            const int key = e >> 6, dk = e & 63;
            Kts[dk * 72 + key] = Kraw[e];
        }
        // ---- prefetch next tile's K/V rows into cache during compute ----
        if (kt + 1 < nkt && tid < 64) {
            __builtin_prefetch(gK + (size_t)(64 + tid) * Dc, 0, 3);
            __builtin_prefetch(gV + (size_t)(64 + tid) * Dc, 0, 3);
        }
        __syncthreads();

        // ---- S = Q * K^T (contract over DK=64, two 32-chunks) ----
        FragCD s[4];
#pragma unroll
        for (int j = 0; j < 4; ++j)
#pragma unroll
            for (int g = 0; g < 8; ++g) s[j].f[g] = 0.0f;

#pragma unroll
        for (int kb = 0; kb < 64; kb += 32) {
            FragAB qf;
            const unsigned short* qrow = &Qs[(w * 16 + lm) * 64 + kb];
            qf.q[0] = *(const uint4*)(qrow + 8 * half);
            qf.q[1] = *(const uint4*)(qrow + 16 + 8 * half);
#pragma unroll
            for (int j = 0; j < 4; ++j) {
                FragAB kf;
                const unsigned short* krow = &Kraw[(j * 16 + lm) * 64 + kb];
                kf.q[0] = *(const uint4*)(krow + 8 * half);
                kf.q[1] = *(const uint4*)(krow + 16 + 8 * half);
                s[j].v = wmma_bf16(qf, kf, s[j].v);
            }
        }

        // ---- scale + causal mask + online softmax ----
        float nmax[8];
#pragma unroll
        for (int g = 0; g < 8; ++g) nmax[g] = rmax[g];
#pragma unroll
        for (int j = 0; j < 4; ++j) {
            const int acol = key0 + j * 16 + lm;
#pragma unroll
            for (int g = 0; g < 8; ++g) {
                const int arow = qbase + w * 16 + g + 8 * half;
                float v = s[j].f[g] * 0.125f;               // 1/sqrt(64)
                if (acol > arow) v = -3.0e38f;
                s[j].f[g] = v;
                nmax[g] = fmaxf(nmax[g], v);
            }
        }
#pragma unroll
        for (int g = 0; g < 8; ++g)
#pragma unroll
            for (int m = 1; m < 16; m <<= 1)
                nmax[g] = fmaxf(nmax[g], __shfl_xor(nmax[g], m, 32));

        float alpha[8], psum[8];
#pragma unroll
        for (int g = 0; g < 8; ++g) {
            alpha[g] = __expf(rmax[g] - nmax[g]);
            rmax[g]  = nmax[g];
            psum[g]  = 0.0f;
        }
#pragma unroll
        for (int j = 0; j < 4; ++j)
#pragma unroll
            for (int g = 0; g < 8; ++g) {
                const float p = __expf(s[j].f[g] - rmax[g]);
                s[j].f[g] = p;
                psum[g] += p;
            }
#pragma unroll
        for (int g = 0; g < 8; ++g)
#pragma unroll
            for (int m = 1; m < 16; m <<= 1)
                psum[g] += __shfl_xor(psum[g], m, 32);
#pragma unroll
        for (int g = 0; g < 8; ++g) rsum[g] = rsum[g] * alpha[g] + psum[g];
#pragma unroll
        for (int jd = 0; jd < 4; ++jd)
#pragma unroll
            for (int g = 0; g < 8; ++g) o[jd].f[g] *= alpha[g];

        // ---- P: C/D layout -> A layout via per-wave LDS round-trip ----
        unsigned short* pw = &Ps[w * 16 * 64];
#pragma unroll
        for (int j = 0; j < 4; ++j)
#pragma unroll
            for (int g = 0; g < 8; ++g)
                pw[(g + 8 * half) * 64 + j * 16 + lm] = f2bf(s[j].f[g]);
        __syncthreads();                    // make P visible/ordered

        // ---- O += P * V (contract over 64 keys, two 32-chunks) ----
#pragma unroll
        for (int koff = 0; koff < 64; koff += 32) {
            FragAB pf;
            const unsigned short* prow = &pw[lm * 64 + koff];
            pf.q[0] = *(const uint4*)(prow + 8 * half);
            pf.q[1] = *(const uint4*)(prow + 16 + 8 * half);
#pragma unroll
            for (int jd = 0; jd < 4; ++jd) {
                FragAB vf_;
                const unsigned short* vrow = &Vs[koff * 64 + jd * 16 + lm];
#pragma unroll
                for (int s2 = 0; s2 < 16; ++s2)
                    vf_.v[s2] = __builtin_bit_cast(__bf16, vrow[((s2 < 8) ? (8 * half + s2) : (16 + 8 * half + s2 - 8)) * 64]);
                o[jd].v = wmma_bf16(pf, vf_, o[jd].v);
            }
        }
    }

    // ---- normalize and store context (bf16, [B,S,H*DK]) ----
    float rinv[8];
#pragma unroll
    for (int g = 0; g < 8; ++g) rinv[g] = 1.0f / rsum[g];
#pragma unroll
    for (int jd = 0; jd < 4; ++jd)
#pragma unroll
        for (int g = 0; g < 8; ++g) {
            const int arow = qbase + w * 16 + g + 8 * half;
            const int c    = h * 64 + jd * 16 + lm;
            ctx[((size_t)b * Sc + arow) * Dc + c] = f2bf(o[jd].f[g] * rinv[g]);
        }
}

// ---------------------------------------------------------------------------
// Launch: QKV projections -> attention -> output projection.
// Workspace (bf16): Q,K,V,ctx each B*S*D = 4,194,304 elements (8 MB each).
// ---------------------------------------------------------------------------
extern "C" void kernel_launch(void* const* d_in, const int* in_sizes, int n_in,
                              void* d_out, int out_size, void* d_ws, size_t ws_size,
                              hipStream_t stream) {
    const float* q  = (const float*)d_in[0];
    const float* k  = (const float*)d_in[1];
    const float* v  = (const float*)d_in[2];
    // d_in[3] = causal mask (known analytically, unused)
    const float* wq = (const float*)d_in[4];
    const float* bq = (const float*)d_in[5];
    const float* wk = (const float*)d_in[6];
    const float* bk = (const float*)d_in[7];
    const float* wv = (const float*)d_in[8];
    const float* bv = (const float*)d_in[9];
    const float* wo = (const float*)d_in[10];
    const float* bo = (const float*)d_in[11];
    float* out = (float*)d_out;

    const size_t elems = (size_t)Bc * Sc * Dc;            // 4,194,304
    unsigned short* Qb = (unsigned short*)d_ws;
    unsigned short* Kb = Qb + elems;
    unsigned short* Vb = Kb + elems;
    unsigned short* Cb = Vb + elems;
    (void)in_sizes; (void)n_in; (void)out_size; (void)ws_size;

    const int M = Bc * Sc, N = Dc, K = Dc;
    dim3 gGemm(M / 128, N / 128);                         // 32 x 8
    dim3 gAttn(Sc / 128, Bc * Hc);                        // 16 x 32

    gemm_wmma<float, false><<<gGemm, 256, 0, stream>>>(q, wq, bq, Qb, M, N, K);
    gemm_wmma<float, false><<<gGemm, 256, 0, stream>>>(k, wk, bk, Kb, M, N, K);
    gemm_wmma<float, false><<<gGemm, 256, 0, stream>>>(v, wv, bv, Vb, M, N, K);

    attn_wmma<<<gAttn, 256, 0, stream>>>(Qb, Kb, Vb, Cb);

    gemm_wmma<unsigned short, true><<<gGemm, 256, 0, stream>>>(Cb, wo, bo, out, M, N, K);
}
